// LSTM_FF_89309549953667
// MI455X (gfx1250) — matrix operand order repaired and checked by reading, not compiled, not run on hardware
//
#include <hip/hip_runtime.h>
#include <stdint.h>

#define B_   64
#define T_   2048
#define H_   512
#define NWG  16
#define NTHR 256
#define KC   128

typedef __attribute__((ext_vector_type(8)))  float  v8f;
typedef __attribute__((ext_vector_type(8)))  __bf16 v8bf;
typedef __attribute__((ext_vector_type(16))) __bf16 v16bf;

#if defined(__gfx1250__) && __has_builtin(__builtin_amdgcn_global_load_async_to_lds_b64) && \
    __has_builtin(__builtin_amdgcn_s_wait_asynccnt)
#define USE_ASYNC_LDS 1
typedef __attribute__((ext_vector_type(2))) int v2i_t;
typedef __attribute__((address_space(1))) v2i_t gas_v2i;   // global 64-bit payload
typedef __attribute__((address_space(3))) v2i_t las_v2i;   // LDS 64-bit payload
#endif

__device__ __forceinline__ unsigned short f2bfu(float f) {
  unsigned u = __builtin_bit_cast(unsigned, f);
  u += 0x7FFFu + ((u >> 16) & 1u);           // round-to-nearest-even
  return (unsigned short)(u >> 16);
}
__device__ __forceinline__ float bf2f(unsigned short h) {
  unsigned u = ((unsigned)h) << 16;
  return __builtin_bit_cast(float, u);
}
__device__ __forceinline__ float sigm(float x) { return 1.f / (1.f + __expf(-x)); }

__device__ __forceinline__ v16bf load_frag(const unsigned short* p0,
                                           const unsigned short* p1) {
  v8bf lo = *(const v8bf*)p0;
  v8bf hi = *(const v8bf*)p1;
  return __builtin_shufflevector(lo, hi, 0,1,2,3,4,5,6,7,8,9,10,11,12,13,14,15);
}

// ---- grid-wide barrier (16 co-resident blocks), agent-scope release/acquire ----
__device__ __forceinline__ void gsync(unsigned int* cnt, unsigned int target) {
  __syncthreads();
  if (threadIdx.x == 0) {
    __threadfence();
    __hip_atomic_fetch_add(cnt, 1u, __ATOMIC_RELEASE, __HIP_MEMORY_SCOPE_AGENT);
    while (__hip_atomic_load(cnt, __ATOMIC_ACQUIRE, __HIP_MEMORY_SCOPE_AGENT) < target)
      __builtin_amdgcn_s_sleep(2);
  }
  __syncthreads();
}

__global__ void init_kernel(unsigned int* cnt) {
  if (threadIdx.x < 8) cnt[threadIdx.x] = 0u;
}

// Pack [W_hh | W_ih] per (layer,dir) into K-major bf16, and combine biases.
__global__ void prep_kernel(
    const float* __restrict__ wih0, const float* __restrict__ whh0,
    const float* __restrict__ bih0, const float* __restrict__ bhh0,
    const float* __restrict__ wih1, const float* __restrict__ whh1,
    const float* __restrict__ bih1, const float* __restrict__ bhh1,
    unsigned short* __restrict__ Wcat0, unsigned short* __restrict__ Wcat1,
    float* __restrict__ bias0, float* __restrict__ bias1)
{
  const size_t tid = (size_t)blockIdx.x * blockDim.x + threadIdx.x;
  const size_t nth = (size_t)gridDim.x * blockDim.x;
  const size_t N0 = 2ull * 2048ull * 768ull;
  for (size_t i = tid; i < N0; i += nth) {
    size_t dn = i / 768, k = i % 768;       // dn = d*2048 + n
    float v = (k < 512) ? whh0[dn * 512 + k] : wih0[dn * 256 + (k - 512)];
    Wcat0[i] = f2bfu(v);
  }
  const size_t N1 = 2ull * 2048ull * 1536ull;
  for (size_t i = tid; i < N1; i += nth) {
    size_t dn = i / 1536, k = i % 1536;
    float v = (k < 512) ? whh1[dn * 512 + k] : wih1[dn * 1024 + (k - 512)];
    Wcat1[i] = f2bfu(v);
  }
  for (size_t i = tid; i < 2ull * 2048ull; i += nth) {
    bias0[i] = bih0[i] + bhh0[i];
    bias1[i] = bih1[i] + bhh1[i];
  }
}

// Persistent grid-synced LSTM recurrence for one (layer, direction).
// Per step: G(64x128 per WG) = [h|x_t](64xKTOT) x Wcat_slice^T via bf16 WMMA,
// then gate nonlinearities + c/h update with length masking.
// Wave tiling: wave w owns M-tile (w&3) and N-half (w>>2): one A fragment
// feeds four back-to-back WMMAs whose B fragments stream from global (L2).
__global__ __launch_bounds__(NTHR) void lstm_recur(
    const void* __restrict__ xsrc, int x_is_f32, int Kin,
    const unsigned short* __restrict__ Wcat, const float* __restrict__ bias,
    unsigned short* __restrict__ hbuf0, unsigned short* __restrict__ hbuf1,
    unsigned short* __restrict__ out0, int dirofs,
    unsigned short* __restrict__ hlast,
    const long long* __restrict__ lens, int rev,
    unsigned int* __restrict__ cnt)
{
  __shared__ __align__(16) unsigned short sA[B_][KC + 8];  // A chunk (padded)
  __shared__ float sG[B_][132];                            // gate staging
  __shared__ float sC[B_][33];                             // cell state slice
  __shared__ float sBias[128];
  __shared__ int   sLen[B_];

  const int tid  = threadIdx.x;
  const int wg   = blockIdx.x;
  const int lane = tid & 31;
  const int wv   = tid >> 5;
  const int mt   = wv & 3;            // M-tile (16 batch rows)
  const int nq   = wv >> 2;           // N-half (64 of 128 local gate cols)
  const int j0   = wg * 32;           // h-column base for this WG
  const int KTOT   = 512 + Kin;
  const int nchunk = KTOT >> 7;       // 768->6, 1536->12

  if (tid < B_)  sLen[tid]  = (int)lens[tid];
  if (tid < 128) sBias[tid] = bias[(tid >> 5) * 512 + j0 + (tid & 31)];
  for (int i = tid; i < B_ * 33; i += NTHR) (&sC[0][0])[i] = 0.f;
  for (int i = tid; i < 2048; i += NTHR) hbuf0[wg * 2048 + i] = 0;  // h0 = 0

  unsigned int phase = 1;
  gsync(cnt, NWG * phase); phase++;

  // per-wave constants for WMMA operand layouts (wave32)
  const int ml   = lane & 15;
  const int kb   = (lane >> 4) << 3;   // A-frag K sub-offset
  const int khiB = (lane >> 4) << 3;   // B-frag K sub-offset
  const int mb   = mt << 4;            // M-tile row base
  // global gate-row (hence weight-row) for each of this wave's 4 N-tiles
  size_t wrow[4];
  #pragma unroll
  for (int nt = 0; nt < 4; ++nt) {
    const int cbase = (nq << 6) + (nt << 4);             // local col base
    const int grow  = (cbase >> 5) * 512 + j0 + (cbase & 31) + ml;
    wrow[nt] = (size_t)grow * (size_t)KTOT;
  }

  for (int t = 0; t < T_; ++t) {
    const unsigned short* __restrict__ hread  = (t & 1) ? hbuf1 : hbuf0;
    unsigned short* __restrict__       hwrite = (t & 1) ? hbuf0 : hbuf1;

    v8f acc[4];
    #pragma unroll
    for (int nt = 0; nt < 4; ++nt)
      #pragma unroll
      for (int v = 0; v < 8; ++v) acc[nt][v] = 0.f;

    for (int kc = 0; kc < nchunk; ++kc) {
      __syncthreads();
      const int k0 = kc << 7;
      // cooperative A-chunk staging: [h | x_t], 64 rows x 128 cols (bf16)
      for (int i = tid; i < B_ * 32; i += NTHR) {
        const int b  = i >> 5;
        const int g4 = (i & 31) << 2;
        const int k  = k0 + g4;
        if (k < 512) {
#ifdef USE_ASYNC_LDS
          __builtin_amdgcn_global_load_async_to_lds_b64(
              (gas_v2i*)(hread + b * H_ + k), (las_v2i*)&sA[b][g4], 0, 0);
#else
          *(unsigned long long*)&sA[b][g4] =
              *(const unsigned long long*)&hread[b * H_ + k];
#endif
        } else {
          int tp = rev ? (sLen[b] - 1 - t) : t;
          tp = tp < 0 ? 0 : (tp > T_ - 1 ? T_ - 1 : tp);
          const int kk = k - 512;
          if (x_is_f32) {
            const float4 f = *(const float4*)((const float*)xsrc +
                               ((size_t)b * T_ + tp) * (size_t)Kin + kk);
            sA[b][g4 + 0] = f2bfu(f.x);
            sA[b][g4 + 1] = f2bfu(f.y);
            sA[b][g4 + 2] = f2bfu(f.z);
            sA[b][g4 + 3] = f2bfu(f.w);
          } else {
            const unsigned short* xp = (const unsigned short*)xsrc +
                ((size_t)b * T_ + tp) * (size_t)Kin + kk;
#ifdef USE_ASYNC_LDS
            __builtin_amdgcn_global_load_async_to_lds_b64(
                (gas_v2i*)xp, (las_v2i*)&sA[b][g4], 0, 0);
#else
            *(unsigned long long*)&sA[b][g4] = *(const unsigned long long*)xp;
#endif
          }
        }
      }
#ifdef USE_ASYNC_LDS
      __builtin_amdgcn_s_wait_asynccnt(0);
#endif
      __syncthreads();
      if (kc + 1 < nchunk)
        __builtin_prefetch(Wcat + wrow[0] + k0 + KC, 0, 0);  // global_prefetch_b8

      // preload this wave's A fragments for all 4 k-steps (one DS clause)
      v16bf afr[4];
      #pragma unroll
      for (int ks = 0; ks < 4; ++ks) {
        const int kloc = ks << 5;
        afr[ks] = load_frag(&sA[mb + ml][kloc + kb], &sA[mb + ml][kloc + 16 + kb]);
      }
      #pragma unroll
      for (int ks = 0; ks < 4; ++ks) {
        const int kofs = k0 + (ks << 5) + khiB;
        // four B fragments (global, L2-resident weights), one shared A
        const unsigned short* bp0 = Wcat + wrow[0] + kofs;
        const unsigned short* bp1 = Wcat + wrow[1] + kofs;
        const unsigned short* bp2 = Wcat + wrow[2] + kofs;
        const unsigned short* bp3 = Wcat + wrow[3] + kofs;
        const v16bf b0 = load_frag(bp0, bp0 + 16);
        const v16bf b1 = load_frag(bp1, bp1 + 16);
        const v16bf b2 = load_frag(bp2, bp2 + 16);
        const v16bf b3 = load_frag(bp3, bp3 + 16);
        acc[0] = __builtin_amdgcn_wmma_f32_16x16x32_bf16(
            false, afr[ks], false, b0, (short)0, acc[0], false, false);
        acc[1] = __builtin_amdgcn_wmma_f32_16x16x32_bf16(
            false, afr[ks], false, b1, (short)0, acc[1], false, false);
        acc[2] = __builtin_amdgcn_wmma_f32_16x16x32_bf16(
            false, afr[ks], false, b2, (short)0, acc[2], false, false);
        acc[3] = __builtin_amdgcn_wmma_f32_16x16x32_bf16(
            false, afr[ks], false, b3, (short)0, acc[3], false, false);
      }
    }

    // stage gates to LDS (C/D layout: lane&15 -> N col, vgpr+8*(lane>>4) -> M row)
    {
      const int rbase = mb + ((lane >> 4) << 3);
      #pragma unroll
      for (int nt = 0; nt < 4; ++nt) {
        const int ncol = (nq << 6) + (nt << 4) + ml;
        #pragma unroll
        for (int v = 0; v < 8; ++v)
          sG[rbase + v][ncol] = acc[nt][v];
      }
    }
    __syncthreads();

    // elementwise gate math + masked state update
    for (int i = tid; i < B_ * 32; i += NTHR) {
      const int b = i >> 5;
      const int j = i & 31;
      const float gi = sG[b][j]      + sBias[j];
      const float gf = sG[b][32 + j] + sBias[32 + j];
      const float gg = sG[b][64 + j] + sBias[64 + j];
      const float go = sG[b][96 + j] + sBias[96 + j];
      const float ii = sigm(gi), ff = sigm(gf), oo = sigm(go);
      const float gc = tanhf(gg);
      const float cprev = sC[b][j];
      const float cn = ff * cprev + ii * gc;
      const float hn = oo * tanhf(cn);
      const bool  msk  = t < sLen[b];
      const float hold = bf2f(hread[b * H_ + j0 + j]);
      const float csel = msk ? cn : cprev;
      const float hsel = msk ? hn : hold;
      sC[b][j] = csel;
      const unsigned short hbits = f2bfu(hsel);
      hwrite[b * H_ + j0 + j] = hbits;
      if (out0) {
        const int tpo = rev ? (sLen[b] - 1 - t) : t;
        if (tpo >= 0)
          out0[((size_t)b * T_ + tpo) * (size_t)(2 * H_) + dirofs + j0 + j] = hbits;
      }
      if (hlast && t == T_ - 1) hlast[b * H_ + j0 + j] = hbits;
    }
    gsync(cnt, NWG * phase); phase++;
  }
}

// Final head: out[b] = sigmoid([h1f|h1b] . w_out + b_out)
__global__ void out_kernel(const unsigned short* __restrict__ hfp,
                           const unsigned short* __restrict__ hbp,
                           const float* __restrict__ w_out,
                           const float* __restrict__ b_out,
                           float* __restrict__ out)
{
  __shared__ float red[NTHR];
  const int b = blockIdx.x, tid = threadIdx.x;
  float s = 0.f;
  for (int k = tid; k < H_; k += NTHR) {
    s += bf2f(hfp[b * H_ + k]) * w_out[k];
    s += bf2f(hbp[b * H_ + k]) * w_out[H_ + k];
  }
  red[tid] = s;
  __syncthreads();
  for (int st = NTHR / 2; st > 0; st >>= 1) {
    if (tid < st) red[tid] += red[tid + st];
    __syncthreads();
  }
  if (tid == 0) out[b] = sigm(red[0] + b_out[0]);
}

extern "C" void kernel_launch(void* const* d_in, const int* in_sizes, int n_in,
                              void* d_out, int out_size, void* d_ws, size_t ws_size,
                              hipStream_t stream)
{
  const float*     seqs = (const float*)d_in[0];
  const long long* lens = (const long long*)d_in[1];
  const float* wih0 = (const float*)d_in[2];
  const float* whh0 = (const float*)d_in[3];
  const float* bih0 = (const float*)d_in[4];
  const float* bhh0 = (const float*)d_in[5];
  const float* wih1 = (const float*)d_in[6];
  const float* whh1 = (const float*)d_in[7];
  const float* bih1 = (const float*)d_in[8];
  const float* bhh1 = (const float*)d_in[9];
  const float* w_out = (const float*)d_in[10];
  const float* b_out = (const float*)d_in[11];

  char* ws = (char*)d_ws;
  size_t off = 0;
  auto take = [&](size_t bytes) -> char* {
    char* p = ws + off;
    off += (bytes + 255) & ~(size_t)255;
    return p;
  };

  unsigned int*   cnt    = (unsigned int*)take(256);
  unsigned short* hbuf0  = (unsigned short*)take((size_t)B_ * H_ * 2);
  unsigned short* hbuf1  = (unsigned short*)take((size_t)B_ * H_ * 2);
  unsigned short* hlastF = (unsigned short*)take((size_t)B_ * H_ * 2);
  unsigned short* hlastB = (unsigned short*)take((size_t)B_ * H_ * 2);
  float*          bias0  = (float*)take(2ull * 2048 * 4);
  float*          bias1  = (float*)take(2ull * 2048 * 4);
  unsigned short* Wcat0  = (unsigned short*)take(2ull * 2048 * 768 * 2);
  unsigned short* Wcat1  = (unsigned short*)take(2ull * 2048 * 1536 * 2);
  unsigned short* out0   = (unsigned short*)take((size_t)B_ * T_ * (2 * H_) * 2);

  init_kernel<<<1, 64, 0, stream>>>(cnt);
  prep_kernel<<<2048, NTHR, 0, stream>>>(wih0, whh0, bih0, bhh0,
                                         wih1, whh1, bih1, bhh1,
                                         Wcat0, Wcat1, bias0, bias1);
  // layer 0 (input = f32 seqs, Kin=256), writes bf16 output sequence
  lstm_recur<<<NWG, NTHR, 0, stream>>>(seqs, 1, 256, Wcat0,
      bias0, hbuf0, hbuf1, out0, 0, nullptr, lens, 0, cnt + 0);
  lstm_recur<<<NWG, NTHR, 0, stream>>>(seqs, 1, 256, Wcat0 + 2048ull * 768,
      bias0 + 2048, hbuf0, hbuf1, out0, 512, nullptr, lens, 1, cnt + 1);
  // layer 1 (input = bf16 layer-0 outputs, Kin=1024), keep only h_last
  lstm_recur<<<NWG, NTHR, 0, stream>>>(out0, 0, 1024, Wcat1,
      bias1, hbuf0, hbuf1, nullptr, 0, hlastF, lens, 0, cnt + 2);
  lstm_recur<<<NWG, NTHR, 0, stream>>>(out0, 0, 1024, Wcat1 + 2048ull * 1536,
      bias1 + 2048, hbuf0, hbuf1, nullptr, 0, hlastB, lens, 1, cnt + 3);
  out_kernel<<<B_, NTHR, 0, stream>>>(hlastF, hlastB, w_out, b_out, (float*)d_out);
}
